// CrossPairMemory_13194139533361
// MI455X (gfx1250) — matrix operand order repaired and checked by reading, not compiled, not run on hardware
//
#include <hip/hip_runtime.h>

// MI455X / gfx1250, wave32. bf16 WMMA GEMM pipeline for CrossPairMemory.
//
// Algebraic refactor: h = concat(pc,mc)@W1 = attn_p@(Vp@W1_top) + attn_m@(Vm@W1_bot)
// -> precompute M (128x3584) per call, turning the 210-GFLOP GEMM into a K=128
// GEMM. Remaining GEMMs (h@W2 105 GFLOP + 28 per-pair GEMMs) run through one
// generic wave32 WMMA bf16 kernel: 64x64x64 tiles, 4 waves/block, double-
// buffered LDS, A-tile via global_load_async_to_lds_b128 (ASYNCcnt), B-tile
// software-pipelined through VGPRs with an LDS transpose.

typedef __attribute__((ext_vector_type(16))) __bf16 v16bf;
typedef __attribute__((ext_vector_type(8)))  float  v8f;

#define NUM_PAIRS 28
#define PAIR_DIM  128
#define MACRO_DIM 256
#define NUM_SLOTS 64
#define BATCH     4096
#define D_ALL     3584   // 28*128
#define D2        7168   // 2*D_ALL

// LDS layout (dynamic shared, starts at LDS offset 0):
//   As[2][64][72] bf16  at offset 0      (per-buffer 64*144 = 9216 B)
//   Bt[2][64][72] bf16  at offset 18432  (transposed: Bt[n][k])
#define ABUF_BYTES 9216
#define BT_BASE    18432
#define SMEM_BYTES 36864

// ---------------------------------------------------------------------------
// f32 -> bf16 elementwise convert
__global__ void cvt_kernel(const float* __restrict__ x, __bf16* __restrict__ y, long n) {
    long i = (long)blockIdx.x * 256 + threadIdx.x;
    if (i < n) y[i] = (__bf16)x[i];
}

// ---------------------------------------------------------------------------
// Per-row: pair-query mean, both 64-slot score dot products, softmax -> bf16
// attn (B,128): cols 0..63 = pair attn, 64..127 = macro attn.
__global__ void attn_kernel(const float* __restrict__ ps, const float* __restrict__ ms,
                            const float* __restrict__ kp, const float* __restrict__ km,
                            __bf16* __restrict__ attnb) {
    const int b = blockIdx.x;
    const int t = threadIdx.x;          // 128 threads
    __shared__ float q[128], qm[256], sc[128], ex[128];
    float s = 0.f;
    for (int p = 0; p < NUM_PAIRS; ++p) s += ps[((long)b * NUM_PAIRS + p) * PAIR_DIM + t];
    q[t] = s * (1.0f / 28.0f);
    qm[t]       = ms[(long)b * MACRO_DIM + t];
    qm[t + 128] = ms[(long)b * MACRO_DIM + 128 + t];
    __syncthreads();
    float sco;
    if (t < 64) {
        const float* kr = kp + t * PAIR_DIM;
        float a = 0.f;
        for (int d = 0; d < PAIR_DIM; ++d) a += q[d] * kr[d];
        sco = a * 0.08838834764831845f;                  // 1/sqrt(128)
    } else {
        const float* kr = km + (t - 64) * MACRO_DIM;
        float a = 0.f;
        for (int d = 0; d < MACRO_DIM; ++d) a += qm[d] * kr[d];
        sco = a * 0.0625f;                               // 1/sqrt(256)
    }
    sc[t] = sco;
    __syncthreads();
    const int base = (t >> 6) << 6;                      // group of 64 (pair / macro)
    float mx = -3.4e38f;
    for (int i = 0; i < 64; ++i) mx = fmaxf(mx, sc[base + i]);
    ex[t] = __expf(sco - mx);
    __syncthreads();
    float sum = 0.f;
    for (int i = 0; i < 64; ++i) sum += ex[base + i];
    attnb[(long)b * 128 + t] = (__bf16)(ex[t] / sum);
}

// ---------------------------------------------------------------------------
// Generic wave32 bf16 WMMA GEMM: C[z] = A[z](M x K) * B[z](K x N) (+ bias[n]).
// Block = 128 threads (4 waves), 64x64 output tile, BK=64 (two x32 WMMA steps).
// M, N set by grid (gridDim.y*64, gridDim.x*64); all dims multiples of 64.
// Double-buffered LDS; A tile copied global->LDS asynchronously.
__global__ void __launch_bounds__(128)
gemm_bf16_kernel(const __bf16* __restrict__ A, const __bf16* __restrict__ Bm,
                 const float* __restrict__ bias,
                 float* __restrict__ Cf, __bf16* __restrict__ Cb,
                 int K, long lda, long ldb, long ldc,
                 long sA, long sB, long sC, long sBias) {
    extern __shared__ __align__(16) char smem[];
    __bf16 (*As)[64][72] = (__bf16 (*)[64][72])(smem);
    __bf16 (*Bt)[64][72] = (__bf16 (*)[64][72])(smem + BT_BASE);

    const int tid  = threadIdx.x;
    const int wv   = tid >> 5;        // wave id 0..3 -> rows 16*wv..16*wv+15
    const int lane = tid & 31;
    const int half = lane >> 4;       // lane group (ISA 7.12.2 layouts)
    const int lm   = lane & 15;
    const v8f zero = {0.f, 0.f, 0.f, 0.f, 0.f, 0.f, 0.f, 0.f};
    v8f acc[4] = {zero, zero, zero, zero};

    const long abase = (long)blockIdx.z * sA + (long)(blockIdx.y * 64) * lda;
    const long bbase = (long)blockIdx.z * sB + (long)(blockIdx.x * 64);

    // Issue async global->LDS copies for the A tile at k0 into buffer nb.
    auto issueA = [&](int k0, int nb) {
#pragma unroll
        for (int i = 0; i < 4; ++i) {
            const int gq = tid + i * 128;              // 512 uint4 quads total
            const int row = gq >> 3, c8 = gq & 7;
            const unsigned lds = (unsigned)(nb * ABUF_BYTES + row * 144 + c8 * 16);
            const void* gp = (const void*)(A + abase + (long)row * lda + (k0 + c8 * 8));
            asm volatile("global_load_async_to_lds_b128 %0, %1, off"
                         :: "v"(lds), "v"((unsigned long long)gp)
                         : "memory");
        }
    };
    // Pipelined B tile: global->VGPR quads, then transpose-store to LDS.
    auto loadBquads = [&](int k0, uint4* bv) {
#pragma unroll
        for (int i = 0; i < 4; ++i) {
            const int gq = tid + i * 128;
            const int kk = gq >> 3, n8 = gq & 7;
            bv[i] = *(const uint4*)(Bm + bbase + (long)(k0 + kk) * ldb + n8 * 8);
        }
    };
    auto storeBt = [&](const uint4* bv, int nb) {
#pragma unroll
        for (int i = 0; i < 4; ++i) {
            const int gq = tid + i * 128;
            const int kk = gq >> 3, n8 = gq & 7;
            const __bf16* e = (const __bf16*)&bv[i];
#pragma unroll
            for (int j = 0; j < 8; ++j) Bt[nb][n8 * 8 + j][kk] = e[j];
        }
    };

    uint4 bv[4];
    // Prologue: tile 0 into buffer 0.
    issueA(0, 0);
    loadBquads(0, bv);
    storeBt(bv, 0);

    const int T = K >> 6;
    for (int t = 0; t < T; ++t) {
        const int nb = t & 1;
        // Drain this wave's async A copies for tile t, then converge. The
        // compiler drains DScnt (our Bt stores) at the barrier itself.
        asm volatile("s_wait_asynccnt 0x0" ::: "memory");
        __syncthreads();
        const bool more = (t + 1) < T;
        if (more) {                       // overlap tile t+1 traffic with compute
            issueA((t + 1) << 6, nb ^ 1);
            loadBquads((t + 1) << 6, bv);
        }
#pragma unroll
        for (int ks = 0; ks < 2; ++ks) {
            const int kb = ks * 32;
            // A fragment (16x32): lanes 0-15: K 0-7 & 16-23; lanes 16-31: K 8-15 & 24-31
            v16bf a;
            {
                const __bf16* r0 = &As[nb][wv * 16 + lm][kb + half * 8];
                const __bf16* r1 = &As[nb][wv * 16 + lm][kb + 16 + half * 8];
#pragma unroll
                for (int j = 0; j < 8; ++j) { a[j] = r0[j]; a[8 + j] = r1[j]; }
            }
#pragma unroll
            for (int tt = 0; tt < 4; ++tt) {
                // B fragment (32x16): lanes 0-15: K 0-7 & 16-23; lanes 16-31: K 8-15 & 24-31
                v16bf bb;
                const __bf16* r0b = &Bt[nb][tt * 16 + lm][kb + half * 8];
                const __bf16* r1b = &Bt[nb][tt * 16 + lm][kb + 16 + half * 8];
#pragma unroll
                for (int j = 0; j < 8; ++j) { bb[j] = r0b[j]; bb[8 + j] = r1b[j]; }
                acc[tt] = __builtin_amdgcn_wmma_f32_16x16x32_bf16(
                    false, a, false, bb, (short)0, acc[tt], false, false);
            }
        }
        if (more) storeBt(bv, nb ^ 1);    // loadcnt wait lands after the WMMAs
    }
    // Epilogue: C VGPR r -> row 8*half + r, col = lane&15
#pragma unroll
    for (int t = 0; t < 4; ++t) {
        const int n = blockIdx.x * 64 + t * 16 + lm;
        float bvs = 0.f;
        if (bias) bvs = bias[(long)blockIdx.z * sBias + n];
#pragma unroll
        for (int r = 0; r < 8; ++r) {
            const int m = blockIdx.y * 64 + wv * 16 + half * 8 + r;
            const float val = acc[t][r] + bvs;
            const long off = (long)blockIdx.z * sC + (long)m * ldc + n;
            if (Cb) Cb[off] = (__bf16)val;
            else    Cf[off] = val;
        }
    }
}

// ---------------------------------------------------------------------------
// Row LayerNorm (dim 3584) + exact GELU -> bf16
__global__ void ln_gelu_kernel(const float* __restrict__ H, const float* __restrict__ g,
                               const float* __restrict__ b, __bf16* __restrict__ out) {
    const int row = blockIdx.x, t = threadIdx.x;   // 256 threads
    const float* x = H + (long)row * D_ALL;
    float s = 0.f, s2 = 0.f;
    for (int j = t; j < D_ALL; j += 256) { float v = x[j]; s += v; s2 += v * v; }
    __shared__ float r1[256], r2[256];
    r1[t] = s; r2[t] = s2; __syncthreads();
    for (int off = 128; off > 0; off >>= 1) {
        if (t < off) { r1[t] += r1[t + off]; r2[t] += r2[t + off]; }
        __syncthreads();
    }
    const float mu = r1[0] * (1.0f / D_ALL);
    const float var = r2[0] * (1.0f / D_ALL) - mu * mu;
    const float rs = rsqrtf(var + 1e-5f);
    __bf16* o = out + (long)row * D_ALL;
    for (int j = t; j < D_ALL; j += 256) {
        const float y = (x[j] - mu) * rs * g[j] + b[j];
        o[j] = (__bf16)(0.5f * y * (1.0f + erff(y * 0.70710678118654752f)));
    }
}

// ---------------------------------------------------------------------------
// Build Xcat (B, 28, 256) bf16 = [pair_states | fused]
__global__ void pack_kernel(const float* __restrict__ ps, const __bf16* __restrict__ fused,
                            __bf16* __restrict__ xcat, long n) {
    long i = (long)blockIdx.x * 256 + threadIdx.x;
    if (i >= n) return;
    const long bp = i >> 7;                 // b*28 + p
    const int  d  = (int)(i & 127);
    const long b  = bp / NUM_PAIRS;
    const int  p  = (int)(bp % NUM_PAIRS);
    xcat[bp * 256 + d]       = (__bf16)ps[i];
    xcat[bp * 256 + 128 + d] = fused[b * (long)D_ALL + p * PAIR_DIM + d];
}

// ---------------------------------------------------------------------------
// Final per-pair LayerNorm (dim 128), in place on d_out
__global__ void final_ln_kernel(float* __restrict__ Y, const float* __restrict__ g,
                                const float* __restrict__ b) {
    const int row = blockIdx.x, t = threadIdx.x;   // 128 threads
    const int p = row % NUM_PAIRS;
    float* x = Y + (long)row * PAIR_DIM;
    const float v = x[t];
    __shared__ float r1[128], r2[128];
    r1[t] = v; r2[t] = v * v; __syncthreads();
    for (int off = 64; off > 0; off >>= 1) {
        if (t < off) { r1[t] += r1[t + off]; r2[t] += r2[t + off]; }
        __syncthreads();
    }
    const float mu = r1[0] * (1.0f / 128.0f);
    const float var = r2[0] * (1.0f / 128.0f) - mu * mu;
    const float rs = rsqrtf(var + 1e-5f);
    x[t] = (v - mu) * rs * g[p * PAIR_DIM + t] + b[p * PAIR_DIM + t];
}

// ---------------------------------------------------------------------------
extern "C" void kernel_launch(void* const* d_in, const int* in_sizes, int n_in,
                              void* d_out, int out_size, void* d_ws, size_t ws_size,
                              hipStream_t stream) {
    (void)in_sizes; (void)n_in; (void)out_size; (void)ws_size;
    const float* ps  = (const float*)d_in[0];
    const float* ms  = (const float*)d_in[1];
    const float* kp  = (const float*)d_in[2];
    const float* vp  = (const float*)d_in[3];
    const float* km  = (const float*)d_in[4];
    const float* vm  = (const float*)d_in[5];
    const float* w1  = (const float*)d_in[6];
    const float* b1  = (const float*)d_in[7];
    const float* lng = (const float*)d_in[8];
    const float* lnb = (const float*)d_in[9];
    const float* w2  = (const float*)d_in[10];
    const float* b2  = (const float*)d_in[11];
    const float* pw  = (const float*)d_in[12];
    const float* pb  = (const float*)d_in[13];
    const float* plg = (const float*)d_in[14];
    const float* plb = (const float*)d_in[15];

    char* ws = (char*)d_ws;
    // region [0, 77MB): W1b+W2b early, aliased by Xcat after both are dead
    __bf16* W1b  = (__bf16*)(ws + 0);            // 7168*3584*2 = 51,380,224
    __bf16* W2b  = (__bf16*)(ws + 51380224);     // 3584*3584*2 = 25,690,112
    __bf16* Xcat = (__bf16*)(ws + 0);            // 4096*7168*2 = 58,720,256 (alias)
    __bf16* Vpb  = (__bf16*)(ws + 77070336);     // 458,752
    __bf16* Vmb  = (__bf16*)(ws + 77529088);     // 458,752
    __bf16* PWb  = (__bf16*)(ws + 77987840);     // 1,835,008
    float*  Mf   = (float*) (ws + 79822848);     // 128*3584*4 = 1,835,008
    __bf16* Mb   = (__bf16*)(ws + 81657856);     // 917,504
    __bf16* Ab   = (__bf16*)(ws + 82575360);     // 4096*128*2 = 1,048,576
    float*  Hraw = (float*) (ws + 83623936);     // 4096*3584*4 = 58,720,256
    __bf16* Hb   = (__bf16*)(ws + 142344192);    // 29,360,128
    __bf16* Fb   = (__bf16*)(ws + 171704320);    // 29,360,128 -> end 201,064,448

    auto cvt = [&](const float* x, __bf16* y, long n) {
        cvt_kernel<<<(unsigned)((n + 255) / 256), 256, 0, stream>>>(x, y, n);
    };
    cvt(w1, W1b, (long)D2 * D_ALL);
    cvt(w2, W2b, (long)D_ALL * D_ALL);
    cvt(vp, Vpb, (long)NUM_SLOTS * D_ALL);
    cvt(vm, Vmb, (long)NUM_SLOTS * D_ALL);
    cvt(pw, PWb, (long)NUM_PAIRS * 256 * 128);

    // M_top = Vp @ W1[0:3584,:]  ; M_bot = Vm @ W1[3584:,:]   (each 64 x 3584)
    gemm_bf16_kernel<<<dim3(D_ALL / 64, 1, 1), 128, SMEM_BYTES, stream>>>(
        Vpb, W1b, nullptr, Mf, nullptr,
        D_ALL, D_ALL, D_ALL, D_ALL, 0, 0, 0, 0);
    gemm_bf16_kernel<<<dim3(D_ALL / 64, 1, 1), 128, SMEM_BYTES, stream>>>(
        Vmb, W1b + (long)D_ALL * D_ALL, nullptr, Mf + (long)64 * D_ALL, nullptr,
        D_ALL, D_ALL, D_ALL, D_ALL, 0, 0, 0, 0);
    cvt(Mf, Mb, 128L * D_ALL);

    // softmax attention weights (B,128) bf16
    attn_kernel<<<BATCH, 128, 0, stream>>>(ps, ms, kp, km, Ab);

    // h = attn @ M + b1   (4096 x 3584, K=128)
    gemm_bf16_kernel<<<dim3(D_ALL / 64, BATCH / 64, 1), 128, SMEM_BYTES, stream>>>(
        Ab, Mb, b1, Hraw, nullptr,
        128, 128, D_ALL, D_ALL, 0, 0, 0, 0);

    // LayerNorm + GELU -> bf16
    ln_gelu_kernel<<<BATCH, 256, 0, stream>>>(Hraw, lng, lnb, Hb);

    // fused = h @ W2 + b2  (4096 x 3584, K=3584) -> bf16
    gemm_bf16_kernel<<<dim3(D_ALL / 64, BATCH / 64, 1), 128, SMEM_BYTES, stream>>>(
        Hb, W2b, b2, nullptr, Fb,
        D_ALL, D_ALL, D_ALL, D_ALL, 0, 0, 0, 0);

    // Xcat = [pair_states | fused]  (B,28,256) bf16
    pack_kernel<<<(unsigned)(((long)BATCH * NUM_PAIRS * 128 + 255) / 256), 256, 0, stream>>>(
        ps, Fb, Xcat, (long)BATCH * NUM_PAIRS * 128);

    // per-pair GEMM (batched over z=28): out[b,p,:] = Xcat[b,p,:] @ pair_w[p] + pair_b[p]
    gemm_bf16_kernel<<<dim3(128 / 64, BATCH / 64, NUM_PAIRS), 128, SMEM_BYTES, stream>>>(
        Xcat, PWb, pb, (float*)d_out, nullptr,
        256, D2, 128, D_ALL,
        /*sA=*/256, /*sB=*/256L * 128, /*sC=*/128, /*sBias=*/128);

    // final per-pair LayerNorm in place on d_out
    final_ln_kernel<<<BATCH * NUM_PAIRS, 128, 0, stream>>>((float*)d_out, plg, plb);
}
